// QSM_V14_47949014893021
// MI455X (gfx1250) — matrix operand order repaired and checked by reading, not compiled, not run on hardware
//
#include <hip/hip_runtime.h>
#include <math.h>

// ---------------------------------------------------------------------------
// Model constants (match reference): V=32000, D=512, H=8, L=4, FF=2048,
// B=4, S=T=512, DK=64.  Tokens per side R = B*S = 2048.
// ---------------------------------------------------------------------------
#define RTOK 2048
#define DM   512
#define NH   8
#define DKH  64
#define FFD  2048
#define VOC  32000
#define SQRT_D 22.62741699796952f

typedef __attribute__((ext_vector_type(16))) __bf16 v16bf;
typedef __attribute__((ext_vector_type(8)))  float  v8f;

union FragBF { uint4 u[2]; v16bf v; };

__device__ __forceinline__ unsigned short f32_bf16(float f) {
  unsigned u = __float_as_uint(f);
  return (unsigned short)((u + 0x7fffu + ((u >> 16) & 1u)) >> 16);  // RNE
}

// issue one 16-byte async global->LDS copy (gfx1250, tracked by ASYNCcnt)
__device__ __forceinline__ void async_copy_b128(const void* lds_ptr,
                                                const void* gptr) {
  unsigned lds_addr = (unsigned)(uintptr_t)lds_ptr;           // LDS byte addr
  unsigned long long ga = (unsigned long long)(uintptr_t)gptr;
  asm volatile("global_load_async_to_lds_b128 %0, %1, off"
               :
               : "v"(lds_addr), "v"(ga)
               : "memory");
}
__device__ __forceinline__ void wait_asynccnt0() {
  asm volatile("s_wait_asynccnt 0x0" ::: "memory");
}

// ---------------------------------------------------------------------------
// fp32 -> bf16(bits) conversion (grid-stride), layout preserved
// ---------------------------------------------------------------------------
__global__ void conv_bf16_kernel(const float* __restrict__ in,
                                 unsigned short* __restrict__ out, long n) {
  long i  = (long)blockIdx.x * blockDim.x + threadIdx.x;
  long st = (long)gridDim.x * blockDim.x;
  for (; i < n; i += st) out[i] = f32_bf16(in[i]);
}

// ---------------------------------------------------------------------------
// fp32 [K][N] -> bf16 [N][K] (transpose + convert), LDS-tiled 32x32.
// grid = (N/32, K/32), 256 threads.
// ---------------------------------------------------------------------------
__global__ void convT_bf16_kernel(const float* __restrict__ in,
                                  unsigned short* __restrict__ out,
                                  int K, int N) {
  __shared__ float tile[32][33];
  int nb = blockIdx.x * 32, kb = blockIdx.y * 32;
  int t = threadIdx.x;
  int r = t >> 5, c = t & 31;  // 8 rows x 32 cols per pass
#pragma unroll
  for (int i = 0; i < 4; ++i)
    tile[r + 8 * i][c] = in[(long)(kb + r + 8 * i) * N + nb + c];
  __syncthreads();
#pragma unroll
  for (int i = 0; i < 4; ++i)
    out[(long)(nb + r + 8 * i) * K + kb + c] = f32_bf16(tile[c][r + 8 * i]);
}

// ---------------------------------------------------------------------------
// bf16 WMMA GEMM: C[M,N] = A[M,K] @ B[K,N] + bias[N]
//   A  : bf16 [M][K] row-major
//   Bt : bf16 [N][K]  (B pre-transposed by convT) -> all staging contiguous
// Block: 256 thr = 8 wave32, tile 32(M) x 256(N), K-step 32.
// Staging uses global_load_async_to_lds_b128 (ASYNCcnt), no VGPR round trip.
// Fragments per ISA 7.12.2 16-bit layouts:
//   A lane l15=m: K runs {8*hi..+7, 16+8*hi..+7}  (two b128 LDS reads)
//   B lane l15=n: K runs {8*hi..+7, 16+8*hi..+7}  (two b128 LDS reads)
// ---------------------------------------------------------------------------
__global__ __launch_bounds__(256) void gemm_bf16_kernel(
    const unsigned short* __restrict__ A, const unsigned short* __restrict__ Bt,
    const float* __restrict__ bias, float* __restrict__ C,
    int M, int N, int K) {
  __shared__ __align__(16) unsigned short ldsA[32 * 32];
  __shared__ __align__(16) unsigned short ldsB[256 * 32];

  const int tid  = threadIdx.x;
  const int wave = tid >> 5, lane = tid & 31;
  const int mw = wave >> 2, nw = wave & 3;
  const int hi = lane >> 4, l15 = lane & 15;
  const int m0 = blockIdx.y * 32, n0 = blockIdx.x * 256;

  v8f acc[4];
  for (int t = 0; t < 4; ++t)
    for (int i = 0; i < 8; ++i) acc[t][i] = 0.0f;

  // A tile: 32 rows x 32 bf16 = 128 x 16B chunks, threads 0..127 one each
  const int ar = tid >> 2, ap = (tid & 3) * 8;
  // B tile: 256 rows(n) x 32 bf16 = 1024 x 16B chunks, 4 per thread

  for (int k0 = 0; k0 < K; k0 += 32) {
    if (tid < 128)
      async_copy_b128(&ldsA[ar * 32 + ap],
                      A + (long)(m0 + ar) * K + k0 + ap);
#pragma unroll
    for (int it = 0; it < 4; ++it) {
      int u = tid + it * 256;
      int n = u >> 2, p = (u & 3) * 8;
      const unsigned short* gB = Bt + (long)(n0 + n) * K + k0 + p;
      async_copy_b128(&ldsB[n * 32 + p], gB);
      if (k0 + 32 < K)  // gfx1250 global_prefetch_b8: next K chunk, same row
        __builtin_prefetch(gB + 32, 0, 1);
    }
    wait_asynccnt0();   // own wave's copies landed in LDS
    __syncthreads();    // everyone's copies landed

    FragBF af;
    int ab = (mw * 16 + l15) * 32 + 8 * hi;
    af.u[0] = *reinterpret_cast<const uint4*>(&ldsA[ab]);
    af.u[1] = *reinterpret_cast<const uint4*>(&ldsA[ab + 16]);

#pragma unroll
    for (int t = 0; t < 4; ++t) {
      FragBF bfr;
      int bb = (nw * 64 + t * 16 + l15) * 32 + 8 * hi;
      bfr.u[0] = *reinterpret_cast<const uint4*>(&ldsB[bb]);
      bfr.u[1] = *reinterpret_cast<const uint4*>(&ldsB[bb + 16]);
      acc[t] = __builtin_amdgcn_wmma_f32_16x16x32_bf16(
          false, af.v, false, bfr.v, (short)0, acc[t], false, false);
    }
    __syncthreads();
  }

  // epilogue: D layout (ISA 7.12.2): VGPR i -> M = i + 8*hi, lane l15 -> N
#pragma unroll
  for (int t = 0; t < 4; ++t) {
    int n = n0 + nw * 64 + t * 16 + l15;
    float bv = bias ? bias[n] : 0.0f;
#pragma unroll
    for (int i = 0; i < 8; ++i) {
      int m = m0 + mw * 16 + hi * 8 + i;
      C[(long)m * N + n] = acc[t][i] + bv;
    }
  }
}

// ---------------------------------------------------------------------------
// embedding gather * sqrt(D).  grid = RTOK blocks, 256 threads.
// ---------------------------------------------------------------------------
__global__ void embed_kernel(const float* __restrict__ emb,
                             const int* __restrict__ tok,
                             float* __restrict__ out) {
  int row = blockIdx.x, t = threadIdx.x;
  const float* e = emb + (long)tok[row] * DM;
  out[(long)row * DM + t]       = e[t]       * SQRT_D;
  out[(long)row * DM + t + 256] = e[t + 256] * SQRT_D;
}

// ---------------------------------------------------------------------------
// LayerNorm over last dim (512): out = LN(x + sub) * g + b.  sub may be null.
// ---------------------------------------------------------------------------
__global__ void ln_kernel(const float* __restrict__ x,
                          const float* __restrict__ sub,
                          const float* __restrict__ g,
                          const float* __restrict__ b,
                          float* __restrict__ out) {
  __shared__ float red[256];
  int row = blockIdx.x, t = threadIdx.x;
  long base = (long)row * DM;
  float a0 = x[base + t]       + (sub ? sub[base + t]       : 0.0f);
  float a1 = x[base + t + 256] + (sub ? sub[base + t + 256] : 0.0f);
  red[t] = a0 + a1;
  __syncthreads();
  for (int o = 128; o > 0; o >>= 1) { if (t < o) red[t] += red[t + o]; __syncthreads(); }
  float mean = red[0] * (1.0f / DM);
  __syncthreads();
  float d0 = a0 - mean, d1 = a1 - mean;
  red[t] = d0 * d0 + d1 * d1;
  __syncthreads();
  for (int o = 128; o > 0; o >>= 1) { if (t < o) red[t] += red[t + o]; __syncthreads(); }
  float rs = rsqrtf(red[0] * (1.0f / DM) + 1e-5f);
  out[base + t]       = d0 * rs * g[t]       + b[t];
  out[base + t + 256] = d1 * rs * g[t + 256] + b[t + 256];
}

// ---------------------------------------------------------------------------
// Attention core: one block (256 thr) per (q, h, b).  Q,K,V are [R][512]
// with head h at columns h*64..h*64+63.  SK fixed = 512.
// ---------------------------------------------------------------------------
__global__ void attn_kernel(const float* __restrict__ Q,
                            const float* __restrict__ K,
                            const float* __restrict__ V,
                            float* __restrict__ O,
                            int causal, int use_alibi) {
  __shared__ float qv[DKH];
  __shared__ float sc[512];
  __shared__ float red[256];
  int q = blockIdx.x, h = blockIdx.y, b = blockIdx.z, t = threadIdx.x;
  long qrow = (long)(b * 512 + q) * DM + h * DKH;
  if (t < DKH) qv[t] = Q[qrow + t];
  __syncthreads();

  float slope = exp2f(-(float)(h + 1));  // H=8 -> 2^(-8/H*(h+1))
  for (int kk = t; kk < 512; kk += 256) {
    const float* kp = K + (long)(b * 512 + kk) * DM + h * DKH;
    float s = 0.0f;
    for (int d = 0; d < DKH; ++d) s += qv[d] * kp[d];
    s *= 0.125f;  // 1/sqrt(64)
    if (use_alibi) s += slope * (float)(kk - q);
    if (causal && kk > q) s = -INFINITY;
    sc[kk] = s;
  }
  __syncthreads();

  red[t] = fmaxf(sc[t], sc[t + 256]);
  __syncthreads();
  for (int o = 128; o > 0; o >>= 1) { if (t < o) red[t] = fmaxf(red[t], red[t + o]); __syncthreads(); }
  float mx = red[0];
  __syncthreads();
  float e0 = __expf(sc[t] - mx), e1 = __expf(sc[t + 256] - mx);
  sc[t] = e0; sc[t + 256] = e1;
  red[t] = e0 + e1;
  __syncthreads();
  for (int o = 128; o > 0; o >>= 1) { if (t < o) red[t] += red[t + o]; __syncthreads(); }
  float inv = 1.0f / red[0];
  __syncthreads();

  int dk = t & 63, ch = t >> 6;
  float acc = 0.0f;
  for (int kk = ch * 128; kk < ch * 128 + 128; ++kk)
    acc += sc[kk] * V[(long)(b * 512 + kk) * DM + h * DKH + dk];
  red[t] = acc;
  __syncthreads();
  if (t < 64) {
    float o4 = red[t] + red[t + 64] + red[t + 128] + red[t + 192];
    O[qrow + t] = o4 * inv;
  }
}

// ---------------------------------------------------------------------------
// exact-erf GELU, in place (grid-stride)
// ---------------------------------------------------------------------------
__global__ void gelu_kernel(float* __restrict__ x, long n) {
  long i  = (long)blockIdx.x * blockDim.x + threadIdx.x;
  long st = (long)gridDim.x * blockDim.x;
  for (; i < n; i += st) {
    float v = x[i];
    x[i] = 0.5f * v * (1.0f + erff(v * 0.7071067811865475f));
  }
}

// ---------------------------------------------------------------------------
// quantum rotation mixing, in place per row.
// ---------------------------------------------------------------------------
__global__ void qrot_kernel(float* __restrict__ E,
                            const float* __restrict__ qr,
                            const float* __restrict__ gate) {
  __shared__ float rowb[DM];
  int row = blockIdx.x, t = threadIdx.x;
  long base = (long)row * DM;
  rowb[t] = E[base + t];
  rowb[t + 256] = E[base + t + 256];
  __syncthreads();
  float g = gate[0];
  for (int c = t; c < DM; c += 256) {
    int h = c >> 6, dk = c & 63;
    int prev = h * 64 + ((dk + 63) & 63);
    float qo = rowb[c] * cosf(qr[c]) + rowb[prev] * sinf(qr[c]);
    E[base + c] = g * qo + (1.0f - g) * rowb[c];
  }
}

// ---------------------------------------------------------------------------
// Host orchestration.  Input flattening: jax tree_flatten (dicts sorted by
// key) — see round-0 comment; 179 leaves total.
// ---------------------------------------------------------------------------
extern "C" void kernel_launch(void* const* d_in, const int* in_sizes, int n_in,
                              void* d_out, int out_size, void* d_ws, size_t ws_size,
                              hipStream_t stream) {
  if (n_in < 179) return;
  auto f = [&](int i) -> const float* { return (const float*)d_in[i]; };
  const int* src = (const int*)d_in[177];
  const int* tgt = (const int*)d_in[178];

  char* ws = (char*)d_ws; size_t off = 0;
  auto alloc = [&](size_t bytes) -> void* {
    void* p = ws + off; off += (bytes + 255) & ~(size_t)255; return p;
  };
  float* x   = (float*)alloc((size_t)RTOK * DM * 4);
  float* y   = (float*)alloc((size_t)RTOK * DM * 4);
  float* tmp = (float*)alloc((size_t)RTOK * DM * 4);
  float* qb  = (float*)alloc((size_t)RTOK * DM * 4);
  float* kb  = (float*)alloc((size_t)RTOK * DM * 4);
  float* vb  = (float*)alloc((size_t)RTOK * DM * 4);
  float* att = (float*)alloc((size_t)RTOK * DM * 4);
  float* enc = (float*)alloc((size_t)RTOK * DM * 4);
  float* h1  = (float*)alloc((size_t)RTOK * FFD * 4);
  unsigned short* abuf = (unsigned short*)alloc((size_t)RTOK * FFD * 2);
  unsigned short* wbuf = (unsigned short*)alloc((size_t)DM * VOC * 2);

  auto conv = [&](const float* s, unsigned short* dst, long n) {
    long g = (n + 255) / 256; if (g > 4096) g = 4096;
    conv_bf16_kernel<<<dim3((unsigned)g), 256, 0, stream>>>(s, dst, n);
  };
  // weight fp32 [K][N] -> bf16 [N][K] in wbuf
  auto convW = [&](const float* w, int K, int N) {
    convT_bf16_kernel<<<dim3(N / 32, K / 32), 256, 0, stream>>>(w, wbuf, K, N);
  };
  auto gemm = [&](const unsigned short* A, const unsigned short* Bt,
                  const float* bias, float* C, int M, int N, int K) {
    gemm_bf16_kernel<<<dim3(N / 256, M / 32), 256, 0, stream>>>(A, Bt, bias, C, M, N, K);
  };
  auto ln = [&](const float* xin, const float* sub, const float* gg,
                const float* bb, float* outp) {
    ln_kernel<<<RTOK, 256, 0, stream>>>(xin, sub, gg, bb, outp);
  };
  // attn param dict base pb: +0 bk +1 bo +2 bq +3 bv +4 wk +5 wo +6 wq +7 wv
  auto run_attn = [&](const float* qsrc, const float* kvsrc, int pb,
                      float* outp, int causal, int alibi) {
    conv(kvsrc, abuf, (long)RTOK * DM);
    convW(f(pb + 4), DM, DM);
    gemm(abuf, wbuf, f(pb + 0), kb, RTOK, DM, DM);           // K
    convW(f(pb + 7), DM, DM);
    gemm(abuf, wbuf, f(pb + 3), vb, RTOK, DM, DM);           // V
    if (qsrc != kvsrc) conv(qsrc, abuf, (long)RTOK * DM);
    convW(f(pb + 6), DM, DM);
    gemm(abuf, wbuf, f(pb + 2), qb, RTOK, DM, DM);           // Q
    attn_kernel<<<dim3(512, NH, 4), 256, 0, stream>>>(qb, kb, vb, att, causal, alibi);
    conv(att, abuf, (long)RTOK * DM);
    convW(f(pb + 5), DM, DM);
    gemm(abuf, wbuf, f(pb + 1), outp, RTOK, DM, DM);         // O proj
  };
  // ffn dict base pb: +0 b1 +1 b2 +2 w1 +3 w2
  auto run_ffn = [&](const float* xin, int pb, float* outp) {
    conv(xin, abuf, (long)RTOK * DM);
    convW(f(pb + 2), DM, FFD);
    gemm(abuf, wbuf, f(pb + 0), h1, RTOK, FFD, DM);
    gelu_kernel<<<4096, 256, 0, stream>>>(h1, (long)RTOK * FFD);
    conv(h1, abuf, (long)RTOK * FFD);
    convW(f(pb + 3), FFD, DM);
    gemm(abuf, wbuf, f(pb + 1), outp, RTOK, DM, FFD);
  };

  // ---------------- encoder ----------------
  embed_kernel<<<RTOK, 256, 0, stream>>>(f(106), src, x);
  for (int l = 0; l < 4; ++l) {
    int base = 107 + 16 * l;
    run_attn(x, x, base + 0, tmp, /*causal=*/0, /*alibi=*/1);
    ln(x, tmp, f(base + 13), f(base + 12), x);               // n1
    run_ffn(x, base + 8, tmp);
    ln(x, tmp, f(base + 15), f(base + 14), x);               // n2
  }
  ln(x, nullptr, f(172), f(171), enc);                       // enc_norm
  qrot_kernel<<<RTOK, 256, 0, stream>>>(enc, f(176), f(175));

  // ---------------- decoder ----------------
  embed_kernel<<<RTOK, 256, 0, stream>>>(f(106), tgt, y);
  for (int l = 0; l < 4; ++l) {
    int base = 26 * l;
    run_attn(y, y, base + 18, tmp, /*causal=*/1, /*alibi=*/1);   // self
    ln(y, tmp, f(base + 13), f(base + 12), y);               // n1
    run_attn(y, enc, base + 0, tmp, /*causal=*/0, /*alibi=*/0);  // cross
    ln(y, tmp, f(base + 15), f(base + 14), y);               // n2
    run_ffn(y, base + 8, tmp);
    ln(y, tmp, f(base + 17), f(base + 16), y);               // n3
  }
  ln(y, nullptr, f(105), f(104), y);                         // dec_norm

  // ---------------- logits: (2048 x 512) @ (512 x 32000) + out_b ----------
  conv(y, abuf, (long)RTOK * DM);
  convW(f(174), DM, VOC);
  gemm(abuf, wbuf, f(173), (float*)d_out, RTOK, VOC, DM);
}
